// PhyloEGNN_36739150250410
// MI455X (gfx1250) — compile-verified, matrix operand order checked by reading, not run on hardware
//
#include <hip/hip_runtime.h>

// ---------------- types ----------------
typedef __attribute__((ext_vector_type(16))) __bf16 bf16x16;
typedef __attribute__((ext_vector_type(8)))  __bf16 bf16x8;
typedef __attribute__((ext_vector_type(8)))  float  f32x8;

#define NN   50000
#define NE   100000
#define NB   64
#define HD   128
#define OUTD 256

static __device__ __forceinline__ float siluf(float x){ return x / (1.f + __expf(-x)); }
static __device__ __forceinline__ float sigm (float x){ return 1.f / (1.f + __expf(-x)); }

static __device__ __forceinline__ f32x8 wmma_bf16(bf16x16 a, bf16x16 b, f32x8 c) {
  // D(f32 16x16) = A(bf16 16x32) * B(bf16 32x16) + C
  return __builtin_amdgcn_wmma_f32_16x16x32_bf16(false, a, false, b, (short)0, c, false, false);
}

// A fragment (16x32 bf16): lane L holds row M=L&15; half h=L>>4 selects K groups.
// element e: K = kbase + (e>>3)*16 + h*8 + (e&7)  -> two contiguous 8-elem runs.
static __device__ __forceinline__ bf16x16 afrag_from_lds(const __bf16* rowp /* row base + kbase + h*8 */) {
  union { bf16x16 v; bf16x8 h[2]; } u;
  u.h[0] = *(const bf16x8*)(rowp);
  u.h[1] = *(const bf16x8*)(rowp + 16);
  return u.v;
}

// ---------------- weight pre-pack into B-fragment order ----------------
// packed[frag*512 + lane*16 + e] where frag = kt*ntiles + nt,
// K = kt*32 + (e>>3)*16 + (lane>>4)*8 + (e&7), N = nt*16 + (lane&15)
__global__ void pack_b_kernel(const float* __restrict__ W, __bf16* __restrict__ out,
                              int K, int Nw, int NwPad)
{
  int ntiles = NwPad >> 4;
  int total  = (K >> 5) * ntiles * 512;
  int tid = blockIdx.x * blockDim.x + threadIdx.x;
  if (tid >= total) return;
  int f = tid >> 9, r = tid & 511;
  int lane = r >> 4, e = r & 15;
  int kt = f / ntiles, nt = f - kt * ntiles;
  int h  = lane >> 4;
  int k  = kt*32 + ((e>>3)<<4) + h*8 + (e&7);
  int n  = nt*16 + (lane & 15);
  float v = (n < Nw) ? W[k*Nw + n] : 0.f;
  out[tid] = (__bf16)v;
}

// ---------------- small utility kernels ----------------
__global__ void copy_f32_kernel(const float* __restrict__ s, float* __restrict__ d, int n){
  int i = blockIdx.x*blockDim.x + threadIdx.x; if (i < n) d[i] = s[i];
}
__global__ void zero_f32_kernel(float* __restrict__ d, int n){
  int i = blockIdx.x*blockDim.x + threadIdx.x; if (i < n) d[i] = 0.f;
}

// ---------------- projection + LayerNorm ----------------
__global__ __launch_bounds__(128)
void proj_ln_kernel(const float* __restrict__ x, const float* __restrict__ Wp,
                    const float* __restrict__ bp, const float* __restrict__ g,
                    const float* __restrict__ b, float* __restrict__ h,
                    __bf16* __restrict__ h_bf, int n)
{
  int node = blockIdx.x; int c = threadIdx.x;
  if (node >= n) return;
  __shared__ float red[128];
  __shared__ float xs[8];
  if (c < 8) xs[c] = x[(size_t)node*8 + c];
  __syncthreads();
  float acc = bp[c];
  for (int k = 0; k < 8; ++k) acc += xs[k] * Wp[k*HD + c];
  red[c] = acc; __syncthreads();
  for (int s = 64; s > 0; s >>= 1){ if (c < s) red[c] += red[c+s]; __syncthreads(); }
  float mu = red[0] * (1.f/HD); __syncthreads();
  float d = acc - mu; red[c] = d*d; __syncthreads();
  for (int s = 64; s > 0; s >>= 1){ if (c < s) red[c] += red[c+s]; __syncthreads(); }
  float inv = rsqrtf(red[0]*(1.f/HD) + 1e-5f);
  float y = d*inv*g[c] + b[c];
  h[(size_t)node*HD + c] = y;
  h_bf[(size_t)node*HD + c] = (__bf16)y;
}

// ---------------- EGCL phase 1: coord MLP + pos update ----------------
// block = 128 threads (4 waves), 64 edges/block, 16 edges per wave tile
__global__ __launch_bounds__(128)
void egcl_coord_kernel(const int* __restrict__ row, const int* __restrict__ col,
                       const __bf16* __restrict__ h_bf,
                       const float* __restrict__ pos_old, float* __restrict__ pos_next,
                       const __bf16* __restrict__ w1p, const float* __restrict__ w1_rel,
                       const float* __restrict__ b1,
                       const __bf16* __restrict__ w2p, const float* __restrict__ b2,
                       const float* __restrict__ ew_w, const float* __restrict__ ew_b,
                       const float* __restrict__ cscale, int nE)
{
  __shared__ __bf16 f_lds[64*256];
  __shared__ __bf16 u_lds[64*256];
  __shared__ int    rI[64], cI[64];
  __shared__ float  rel_lds[64*3];

  const int tid = threadIdx.x, lane = tid & 31, wave = tid >> 5;
  const int e0 = blockIdx.x * 64;

  if (tid < 64) {
    int e = e0 + tid;
    int r = (e < nE) ? row[e] : 0;
    int c = (e < nE) ? col[e] : 0;
    rI[tid] = r; cI[tid] = c;
    float rx=0.f, ry=0.f, rz=0.f;
    if (e < nE) {
      rx = pos_old[(size_t)r*3+0] - pos_old[(size_t)c*3+0];
      ry = pos_old[(size_t)r*3+1] - pos_old[(size_t)c*3+1];
      rz = pos_old[(size_t)r*3+2] - pos_old[(size_t)c*3+2];
    }
    rel_lds[tid*3+0]=rx; rel_lds[tid*3+1]=ry; rel_lds[tid*3+2]=rz;
  }
  __syncthreads();

  // stage f = [h[row] | h[col]] as bf16
  for (int c8 = tid; c8 < 64*32; c8 += 128) {
    int m = c8 >> 5, k8 = (c8 & 31)*8;
    int e = e0 + m;
    bf16x8 v;
    if (e < nE) {
      int src = (k8 < HD) ? rI[m] : cI[m];
      v = *(const bf16x8*)(h_bf + (size_t)src*HD + (k8 & (HD-1)));
    } else { for (int i=0;i<8;i++) v[i] = (__bf16)0.f; }
    *(bf16x8*)(f_lds + m*256 + k8) = v;
  }
  __syncthreads();

  const int mrow = lane & 15, half = lane >> 4;
  const __bf16* myrow = f_lds + (wave*16 + mrow)*256;
  bf16x16 af[8];
  for (int kt = 0; kt < 8; ++kt) af[kt] = afrag_from_lds(myrow + kt*32 + half*8);

  float rx[8], ry[8], rz[8];
  for (int v = 0; v < 8; ++v) {
    int m = wave*16 + v + 8*half;
    rx[v]=rel_lds[m*3+0]; ry[v]=rel_lds[m*3+1]; rz[v]=rel_lds[m*3+2];
  }

  // GEMM1: u = silu(f @ coord1 + b1)   (K=256 via WMMA, rel rows in epilogue)
  for (int nt = 0; nt < 16; ++nt) {
    f32x8 acc = {};
    for (int kt = 0; kt < 8; ++kt) {
      bf16x16 bf = *(const bf16x16*)(w1p + ((size_t)(kt*16 + nt))*512 + lane*16);
      acc = wmma_bf16(af[kt], bf, acc);
    }
    int nH = nt*16 + (lane & 15);
    float bias = b1[nH];
    float wr0 = w1_rel[0*256 + nH], wr1 = w1_rel[1*256 + nH], wr2 = w1_rel[2*256 + nH];
    for (int v = 0; v < 8; ++v) {
      int m = wave*16 + v + 8*half;
      float val = acc[v] + bias + rx[v]*wr0 + ry[v]*wr1 + rz[v]*wr2;
      u_lds[m*256 + nH] = (__bf16)siluf(val);
    }
  }
  __syncthreads();

  // GEMM2: d = u @ coord2 (N padded 3->16)
  const __bf16* myrow2 = u_lds + (wave*16 + mrow)*256;
  bf16x16 af2[8];
  for (int kt = 0; kt < 8; ++kt) af2[kt] = afrag_from_lds(myrow2 + kt*32 + half*8);
  f32x8 acc2 = {};
  for (int kt = 0; kt < 8; ++kt) {
    bf16x16 bf = *(const bf16x16*)(w2p + (size_t)kt*512 + lane*16);
    acc2 = wmma_bf16(af2[kt], bf, acc2);
  }

  float scl = cscale[0];
  int nIdx = lane & 15;
  float bAdd = (nIdx < 3) ? b2[nIdx] : 0.f;
  for (int v = 0; v < 8; ++v) {
    float myval = acc2[v] + bAdd;
    int baseLane = half << 4;
    float dx = __shfl(myval, baseLane + 0, 32);
    float dy = __shfl(myval, baseLane + 1, 32);
    float dz = __shfl(myval, baseLane + 2, 32);
    float nrm = sqrtf(dx*dx + dy*dy + dz*dz);
    float s = scl / fmaxf(nrm, 1e-8f);
    float gate = sigm(rx[v]*ew_w[0] + ry[v]*ew_w[1] + rz[v]*ew_w[2] + ew_b[0]);
    int m = wave*16 + v + 8*half;
    int e = e0 + m;
    if (nIdx < 3 && e < nE) {
      atomicAdd(&pos_next[(size_t)rI[m]*3 + nIdx], myval * s * gate);
    }
  }
}

// ---------------- EGCL phase 2: message MLP + aggregation ----------------
__global__ __launch_bounds__(128)
void egcl_msg_kernel(const int* __restrict__ row, const int* __restrict__ col,
                     const __bf16* __restrict__ h_bf, const float* __restrict__ pos_new,
                     float* __restrict__ agg,
                     const __bf16* __restrict__ w1p, const float* __restrict__ w1_rel,
                     const float* __restrict__ b1,
                     const __bf16* __restrict__ w2p, const float* __restrict__ b2, int nE)
{
  __shared__ __bf16 f_lds[64*256];
  __shared__ __bf16 u_lds[64*256];
  __shared__ int    rI[64], cI[64];
  __shared__ float  rel_lds[64*3];

  const int tid = threadIdx.x, lane = tid & 31, wave = tid >> 5;
  const int e0 = blockIdx.x * 64;

  if (tid < 64) {
    int e = e0 + tid;
    int r = (e < nE) ? row[e] : 0;
    int c = (e < nE) ? col[e] : 0;
    rI[tid] = r; cI[tid] = c;
    float rxv=0.f, ryv=0.f, rzv=0.f;
    if (e < nE) {
      rxv = pos_new[(size_t)r*3+0] - pos_new[(size_t)c*3+0];
      ryv = pos_new[(size_t)r*3+1] - pos_new[(size_t)c*3+1];
      rzv = pos_new[(size_t)r*3+2] - pos_new[(size_t)c*3+2];
    }
    rel_lds[tid*3+0]=rxv; rel_lds[tid*3+1]=ryv; rel_lds[tid*3+2]=rzv;
  }
  __syncthreads();

  for (int c8 = tid; c8 < 64*32; c8 += 128) {
    int m = c8 >> 5, k8 = (c8 & 31)*8;
    int e = e0 + m;
    bf16x8 v;
    if (e < nE) {
      int src = (k8 < HD) ? rI[m] : cI[m];
      v = *(const bf16x8*)(h_bf + (size_t)src*HD + (k8 & (HD-1)));
    } else { for (int i=0;i<8;i++) v[i] = (__bf16)0.f; }
    *(bf16x8*)(f_lds + m*256 + k8) = v;
  }
  __syncthreads();

  const int mrow = lane & 15, half = lane >> 4;
  const __bf16* myrow = f_lds + (wave*16 + mrow)*256;
  bf16x16 af[8];
  for (int kt = 0; kt < 8; ++kt) af[kt] = afrag_from_lds(myrow + kt*32 + half*8);

  float rx[8], ry[8], rz[8];
  for (int v = 0; v < 8; ++v) {
    int m = wave*16 + v + 8*half;
    rx[v]=rel_lds[m*3+0]; ry[v]=rel_lds[m*3+1]; rz[v]=rel_lds[m*3+2];
  }

  for (int nt = 0; nt < 16; ++nt) {
    f32x8 acc = {};
    for (int kt = 0; kt < 8; ++kt) {
      bf16x16 bf = *(const bf16x16*)(w1p + ((size_t)(kt*16 + nt))*512 + lane*16);
      acc = wmma_bf16(af[kt], bf, acc);
    }
    int nH = nt*16 + (lane & 15);
    float bias = b1[nH];
    float wr0 = w1_rel[0*256 + nH], wr1 = w1_rel[1*256 + nH], wr2 = w1_rel[2*256 + nH];
    for (int v = 0; v < 8; ++v) {
      int m = wave*16 + v + 8*half;
      float val = acc[v] + bias + rx[v]*wr0 + ry[v]*wr1 + rz[v]*wr2;
      u_lds[m*256 + nH] = (__bf16)siluf(val);
    }
  }
  __syncthreads();

  const __bf16* myrow2 = u_lds + (wave*16 + mrow)*256;
  bf16x16 af2[8];
  for (int kt = 0; kt < 8; ++kt) af2[kt] = afrag_from_lds(myrow2 + kt*32 + half*8);

  for (int nt = 0; nt < 8; ++nt) {
    f32x8 acc = {};
    for (int kt = 0; kt < 8; ++kt) {
      bf16x16 bf = *(const bf16x16*)(w2p + ((size_t)(kt*8 + nt))*512 + lane*16);
      acc = wmma_bf16(af2[kt], bf, acc);
    }
    int nc = nt*16 + (lane & 15);
    float bias = b2[nc];
    for (int v = 0; v < 8; ++v) {
      int m = wave*16 + v + 8*half;
      int e = e0 + m;
      if (e < nE) atomicAdd(&agg[(size_t)rI[m]*HD + nc], siluf(acc[v] + bias));
    }
  }
}

// ---------------- EGCL phase 3: node MLP + residual + LN ----------------
__global__ __launch_bounds__(128)
void egcl_node_kernel(float* __restrict__ h, __bf16* __restrict__ h_bf,
                      const float* __restrict__ agg,
                      const __bf16* __restrict__ w1p, const float* __restrict__ b1,
                      const __bf16* __restrict__ w2p, const float* __restrict__ b2,
                      const float* __restrict__ lng, const float* __restrict__ lnb, int n)
{
  __shared__ __bf16 f_lds[64*256];
  __shared__ __bf16 u_lds[64*256];
  __shared__ float  o_lds[64*128];

  const int tid = threadIdx.x, lane = tid & 31, wave = tid >> 5;
  const int n0 = blockIdx.x * 64;

  // stage f = [h | agg] as bf16
  for (int c8 = tid; c8 < 64*32; c8 += 128) {
    int m = c8 >> 5, k8 = (c8 & 31)*8;
    int node = n0 + m;
    bf16x8 v;
    if (node < n) {
      if (k8 < HD) v = *(const bf16x8*)(h_bf + (size_t)node*HD + k8);
      else {
        const float* a = agg + (size_t)node*HD + (k8 - HD);
        for (int i=0;i<8;i++) v[i] = (__bf16)a[i];
      }
    } else { for (int i=0;i<8;i++) v[i] = (__bf16)0.f; }
    *(bf16x8*)(f_lds + m*256 + k8) = v;
  }
  __syncthreads();

  const int mrow = lane & 15, half = lane >> 4;
  const __bf16* myrow = f_lds + (wave*16 + mrow)*256;
  bf16x16 af[8];
  for (int kt = 0; kt < 8; ++kt) af[kt] = afrag_from_lds(myrow + kt*32 + half*8);

  for (int nt = 0; nt < 16; ++nt) {
    f32x8 acc = {};
    for (int kt = 0; kt < 8; ++kt) {
      bf16x16 bf = *(const bf16x16*)(w1p + ((size_t)(kt*16 + nt))*512 + lane*16);
      acc = wmma_bf16(af[kt], bf, acc);
    }
    int nH = nt*16 + (lane & 15);
    float bias = b1[nH];
    for (int v = 0; v < 8; ++v) {
      int m = wave*16 + v + 8*half;
      u_lds[m*256 + nH] = (__bf16)siluf(acc[v] + bias);
    }
  }
  __syncthreads();

  const __bf16* myrow2 = u_lds + (wave*16 + mrow)*256;
  bf16x16 af2[8];
  for (int kt = 0; kt < 8; ++kt) af2[kt] = afrag_from_lds(myrow2 + kt*32 + half*8);

  for (int nt = 0; nt < 8; ++nt) {
    f32x8 acc = {};
    for (int kt = 0; kt < 8; ++kt) {
      bf16x16 bf = *(const bf16x16*)(w2p + ((size_t)(kt*8 + nt))*512 + lane*16);
      acc = wmma_bf16(af2[kt], bf, acc);
    }
    int nc = nt*16 + (lane & 15);
    float bias = b2[nc];
    for (int v = 0; v < 8; ++v) {
      int m = wave*16 + v + 8*half;
      int node = n0 + m;
      float res = ((node < n) ? h[(size_t)node*HD + nc] : 0.f) + acc[v] + bias;
      o_lds[m*HD + nc] = res;
    }
  }
  __syncthreads();

  if (tid < 64) {
    int node = n0 + tid;
    if (node < n) {
      const float* r = o_lds + tid*HD;
      float mu = 0.f; for (int k=0;k<HD;k++) mu += r[k]; mu *= (1.f/HD);
      float var = 0.f; for (int k=0;k<HD;k++){ float d=r[k]-mu; var += d*d; } var *= (1.f/HD);
      float inv = rsqrtf(var + 1e-5f);
      for (int k=0;k<HD;k++){
        float y = (r[k]-mu)*inv*lng[k] + lnb[k];
        h[(size_t)node*HD + k] = y;
        h_bf[(size_t)node*HD + k] = (__bf16)y;
      }
    }
  }
}

// ---------------- gate MLP ----------------
__global__ __launch_bounds__(128)
void gate_kernel(const __bf16* __restrict__ h_bf,
                 const __bf16* __restrict__ g1p, const float* __restrict__ g1b,
                 const float* __restrict__ lng, const float* __restrict__ lnb,
                 const __bf16* __restrict__ g2p, const float* __restrict__ g2b,
                 const float* __restrict__ g3w, const float* __restrict__ g3b,
                 float* __restrict__ gout, int n)
{
  __shared__ __bf16 f_lds[64*128];
  __shared__ float  o_lds[64*128];
  __shared__ __bf16 u_lds[64*128];
  __shared__ float  q_lds[64*64];

  const int tid = threadIdx.x, lane = tid & 31, wave = tid >> 5;
  const int n0 = blockIdx.x * 64;

  for (int c8 = tid; c8 < 64*16; c8 += 128) {
    int m = c8 >> 4, k8 = (c8 & 15)*8;
    int node = n0 + m;
    bf16x8 v;
    if (node < n) v = *(const bf16x8*)(h_bf + (size_t)node*HD + k8);
    else for (int i=0;i<8;i++) v[i] = (__bf16)0.f;
    *(bf16x8*)(f_lds + m*HD + k8) = v;
  }
  __syncthreads();

  const int mrow = lane & 15, half = lane >> 4;
  const __bf16* myrow = f_lds + (wave*16 + mrow)*HD;
  bf16x16 af[4];
  for (int kt = 0; kt < 4; ++kt) af[kt] = afrag_from_lds(myrow + kt*32 + half*8);

  for (int nt = 0; nt < 8; ++nt) {
    f32x8 acc = {};
    for (int kt = 0; kt < 4; ++kt) {
      bf16x16 bf = *(const bf16x16*)(g1p + ((size_t)(kt*8 + nt))*512 + lane*16);
      acc = wmma_bf16(af[kt], bf, acc);
    }
    int nc = nt*16 + (lane & 15);
    float bias = g1b[nc];
    for (int v = 0; v < 8; ++v) {
      int m = wave*16 + v + 8*half;
      o_lds[m*HD + nc] = acc[v] + bias;
    }
  }
  __syncthreads();

  if (tid < 64) {
    const float* r = o_lds + tid*HD;
    float mu = 0.f; for (int k=0;k<HD;k++) mu += r[k]; mu *= (1.f/HD);
    float var = 0.f; for (int k=0;k<HD;k++){ float d=r[k]-mu; var += d*d; } var *= (1.f/HD);
    float inv = rsqrtf(var + 1e-5f);
    __bf16* uo = u_lds + tid*HD;
    for (int k=0;k<HD;k++){
      float y = (r[k]-mu)*inv*lng[k] + lnb[k];
      uo[k] = (__bf16)fmaxf(y, 0.f);
    }
  }
  __syncthreads();

  const __bf16* myrow2 = u_lds + (wave*16 + mrow)*HD;
  bf16x16 af2[4];
  for (int kt = 0; kt < 4; ++kt) af2[kt] = afrag_from_lds(myrow2 + kt*32 + half*8);

  for (int nt = 0; nt < 4; ++nt) {
    f32x8 acc = {};
    for (int kt = 0; kt < 4; ++kt) {
      bf16x16 bf = *(const bf16x16*)(g2p + ((size_t)(kt*4 + nt))*512 + lane*16);
      acc = wmma_bf16(af2[kt], bf, acc);
    }
    int nc = nt*16 + (lane & 15);
    float bias = g2b[nc];
    for (int v = 0; v < 8; ++v) {
      int m = wave*16 + v + 8*half;
      q_lds[m*64 + nc] = fmaxf(acc[v] + bias, 0.f);
    }
  }
  __syncthreads();

  if (tid < 64) {
    int node = n0 + tid;
    if (node < n) {
      float acc = g3b[0];
      const float* r = q_lds + tid*64;
      for (int k=0;k<64;k++) acc += r[k]*g3w[k];
      gout[node] = acc;
    }
  }
}

// ---------------- softmax pooling ----------------
static __device__ __forceinline__ unsigned f2ord(float f){
  unsigned u = __float_as_uint(f);
  return (u & 0x80000000u) ? ~u : (u | 0x80000000u);
}
static __device__ __forceinline__ float ord2f(unsigned u){
  return (u & 0x80000000u) ? __uint_as_float(u & 0x7FFFFFFFu) : __uint_as_float(~u);
}

__global__ void init_pool_kernel(unsigned* mmax, float* ssum, float* pooled){
  int i = blockIdx.x*blockDim.x + threadIdx.x;
  if (i < NB){ mmax[i] = 0x007FFFFFu; ssum[i] = 0.f; } // ord(-inf)
  if (i < NB*HD) pooled[i] = 0.f;
}
__global__ void seg_max_kernel(const float* __restrict__ g, const int* __restrict__ batch,
                               unsigned* __restrict__ mmax, int n){
  int i = blockIdx.x*blockDim.x + threadIdx.x;
  if (i < n) atomicMax(&mmax[batch[i]], f2ord(g[i]));
}
__global__ void seg_exp_sum_kernel(const float* __restrict__ g, const int* __restrict__ batch,
                                   const unsigned* __restrict__ mmax, float* __restrict__ a,
                                   float* __restrict__ ssum, int n){
  int i = blockIdx.x*blockDim.x + threadIdx.x;
  if (i < n) {
    float m = ord2f(mmax[batch[i]]);
    float e = __expf(g[i] - m);
    a[i] = e;
    atomicAdd(&ssum[batch[i]], e);
  }
}
__global__ void pool_weighted_kernel(const float* __restrict__ h, const float* __restrict__ a,
                                     const int* __restrict__ batch, const float* __restrict__ ssum,
                                     float* __restrict__ pooled, int n){
  int i = blockIdx.x*blockDim.x + threadIdx.x;
  if (i < n*HD) {
    int node = i >> 7, c = i & (HD-1);
    int b = batch[node];
    float w = a[node] / (ssum[b] + 1e-16f);
    atomicAdd(&pooled[b*HD + c], w * h[i]);
  }
}
__global__ void out_kernel(const float* __restrict__ pooled, const float* __restrict__ Wo,
                           const float* __restrict__ bo, float* __restrict__ out){
  int i = blockIdx.x*blockDim.x + threadIdx.x;
  if (i < NB*OUTD) {
    int b = i >> 8, o = i & (OUTD-1);
    float acc = bo[o];
    const float* p = pooled + b*HD;
    for (int k = 0; k < HD; ++k) acc += p[k] * Wo[k*OUTD + o];
    out[i] = acc;
  }
}

// ---------------- host ----------------
static inline int cdiv(int a, int b){ return (a + b - 1) / b; }

extern "C" void kernel_launch(void* const* d_in, const int* in_sizes, int n_in,
                              void* d_out, int out_size, void* d_ws, size_t ws_size,
                              hipStream_t stream)
{
  (void)in_sizes; (void)n_in; (void)out_size; (void)ws_size;

  const float* x_in   = (const float*)d_in[0];
  const float* pos_in = (const float*)d_in[1];
  const int*   row    = (const int*)d_in[2];
  const int*   col    = ((const int*)d_in[2]) + NE;
  const int*   batch  = (const int*)d_in[3];

  // params flattened: top-level dict keys sorted (jax pytree):
  // gate1, gate2, gate3, gate_ln, layers[0..3], out, proj, proj_ln
  const float* gate1_w = (const float*)d_in[4];
  const float* gate1_b = (const float*)d_in[5];
  const float* gate2_w = (const float*)d_in[6];
  const float* gate2_b = (const float*)d_in[7];
  const float* gate3_w = (const float*)d_in[8];
  const float* gate3_b = (const float*)d_in[9];
  const float* gln_g   = (const float*)d_in[10];
  const float* gln_b   = (const float*)d_in[11];
  // per-layer dict keys sorted: coord1(w,b), coord2(w,b), coors_scale,
  // edge1(w,b), edge2(w,b), edge_w(w,b), ln(g,b), node1(w,b), node2(w,b) => 17 leaves
#define LP(l, o) ((const float*)d_in[12 + (l)*17 + (o)])
  const float* out_w  = (const float*)d_in[80];
  const float* out_b  = (const float*)d_in[81];
  const float* proj_w = (const float*)d_in[82];
  const float* proj_b = (const float*)d_in[83];
  const float* pln_g  = (const float*)d_in[84];
  const float* pln_b  = (const float*)d_in[85];

  // workspace carve
  char* wsp = (char*)d_ws;
  auto carve = [&](size_t bytes)->void* {
    void* p = (void*)wsp; wsp += (bytes + 255) & ~((size_t)255); return p;
  };
  float*   h      = (float*)  carve((size_t)NN*HD*4);
  __bf16*  h_bf   = (__bf16*) carve((size_t)NN*HD*2);
  float*   posA   = (float*)  carve((size_t)NN*3*4);
  float*   posB   = (float*)  carve((size_t)NN*3*4);
  float*   agg    = (float*)  carve((size_t)NN*HD*4);
  float*   gbuf   = (float*)  carve((size_t)NN*4);
  float*   abuf   = (float*)  carve((size_t)NN*4);
  unsigned* mmax  = (unsigned*)carve(NB*4);
  float*   ssum   = (float*)  carve(NB*4);
  float*   pooled = (float*)  carve((size_t)NB*HD*4);

  __bf16 *c1p[4], *c2p[4], *e1p[4], *e2p[4], *n1p[4], *n2p[4];
  for (int l = 0; l < 4; ++l) {
    c1p[l] = (__bf16*)carve((size_t)8*16*512*2);
    c2p[l] = (__bf16*)carve((size_t)8*1*512*2);
    e1p[l] = (__bf16*)carve((size_t)8*16*512*2);
    e2p[l] = (__bf16*)carve((size_t)8*8*512*2);
    n1p[l] = (__bf16*)carve((size_t)8*16*512*2);
    n2p[l] = (__bf16*)carve((size_t)8*8*512*2);
  }
  __bf16* g1p = (__bf16*)carve((size_t)4*8*512*2);
  __bf16* g2p = (__bf16*)carve((size_t)4*4*512*2);

  auto packB = [&](const float* W, __bf16* outp, int K, int Nw, int NwPad){
    int total = (K>>5)*(NwPad>>4)*512;
    pack_b_kernel<<<cdiv(total,256), 256, 0, stream>>>(W, outp, K, Nw, NwPad);
  };

  // pack all WMMA-B operands (weights, bf16, fragment order)
  for (int l = 0; l < 4; ++l) {
    packB(LP(l,0),  c1p[l], 256, 256, 256);  // coord1 rows 0..255
    packB(LP(l,2),  c2p[l], 256,   3,  16);  // coord2 padded to N=16
    packB(LP(l,5),  e1p[l], 256, 256, 256);  // edge1 rows 0..255
    packB(LP(l,7),  e2p[l], 256, 128, 128);  // edge2
    packB(LP(l,13), n1p[l], 256, 256, 256);  // node1
    packB(LP(l,15), n2p[l], 256, 128, 128);  // node2
  }
  packB(gate1_w, g1p, 128, 128, 128);
  packB(gate2_w, g2p, 128,  64,  64);

  // initial pos + projection
  copy_f32_kernel<<<cdiv(NN*3,256), 256, 0, stream>>>(pos_in, posA, NN*3);
  proj_ln_kernel<<<NN, 128, 0, stream>>>(x_in, proj_w, proj_b, pln_g, pln_b, h, h_bf, NN);

  float* pcur = posA; float* pnext = posB;
  const int egrid = cdiv(NE, 64);
  const int ngrid = cdiv(NN, 64);

  for (int l = 0; l < 4; ++l) {
    copy_f32_kernel<<<cdiv(NN*3,256), 256, 0, stream>>>(pcur, pnext, NN*3);
    egcl_coord_kernel<<<egrid, 128, 0, stream>>>(row, col, h_bf, pcur, pnext,
        c1p[l], LP(l,0) + 256*256, LP(l,1), c2p[l], LP(l,3),
        LP(l,9), LP(l,10), LP(l,4), NE);
    zero_f32_kernel<<<cdiv(NN*HD,256), 256, 0, stream>>>(agg, NN*HD);
    egcl_msg_kernel<<<egrid, 128, 0, stream>>>(row, col, h_bf, pnext, agg,
        e1p[l], LP(l,5) + 256*256, LP(l,6), e2p[l], LP(l,8), NE);
    egcl_node_kernel<<<ngrid, 128, 0, stream>>>(h, h_bf, agg,
        n1p[l], LP(l,14), n2p[l], LP(l,16), LP(l,11), LP(l,12), NN);
    float* t = pcur; pcur = pnext; pnext = t;
  }

  gate_kernel<<<ngrid, 128, 0, stream>>>(h_bf, g1p, gate1_b, gln_g, gln_b,
                                         g2p, gate2_b, gate3_w, gate3_b, gbuf, NN);
  init_pool_kernel<<<cdiv(NB*HD,256), 256, 0, stream>>>(mmax, ssum, pooled);
  seg_max_kernel<<<cdiv(NN,256), 256, 0, stream>>>(gbuf, batch, mmax, NN);
  seg_exp_sum_kernel<<<cdiv(NN,256), 256, 0, stream>>>(gbuf, batch, mmax, abuf, ssum, NN);
  pool_weighted_kernel<<<cdiv(NN*HD,256), 256, 0, stream>>>(h, abuf, batch, ssum, pooled, NN);
  out_kernel<<<cdiv(NB*OUTD,256), 256, 0, stream>>>(pooled, out_w, out_b, (float*)d_out);
#undef LP
}
